// Coattention_41798621725349
// MI455X (gfx1250) — matrix operand order repaired and checked by reading, not compiled, not run on hardware
//
#include <hip/hip_runtime.h>
#include <hip/hip_bf16.h>

#define BN 32
#define CL 1024
#define QL 128
#define HD 512
#define NEGV (-1e30f)

typedef __attribute__((ext_vector_type(2)))  float  v2f;
typedef __attribute__((ext_vector_type(8)))  float  v8f;
typedef __attribute__((ext_vector_type(16))) __bf16 v16bf;

__device__ __forceinline__ float wave_reduce_sum(float v) {
#pragma unroll
  for (int off = 16; off >= 1; off >>= 1) v += __shfl_xor(v, off, 32);
  return v;
}
__device__ __forceinline__ float wave_reduce_max(float v) {
#pragma unroll
  for (int off = 16; off >= 1; off >>= 1) v = fmaxf(v, __shfl_xor(v, off, 32));
  return v;
}
// K index within a 16-bit A(16x32) fragment: element e of lane-half hi
__device__ __forceinline__ int kmapA16(int e, int hi) {
  return e + 8 * hi + (e < 8 ? 0 : 8);
}

// ---------------- K1: rowC = C.w1 , colQ = Q.w2 (one wave per row) ----------
__global__ __launch_bounds__(256) void k1_dots(
    const float* __restrict__ C, const float* __restrict__ Q,
    const float* __restrict__ w, float* __restrict__ rowC,
    float* __restrict__ colQ) {
  int wave = threadIdx.x >> 5, lane = threadIdx.x & 31;
  int id = blockIdx.x * 8 + wave;
  if (id < BN * CL) {
    const float* row = C + (size_t)id * HD;
    float s = 0.f;
    for (int k = lane; k < HD; k += 32) s += row[k] * w[k];
    s = wave_reduce_sum(s);
    if (lane == 0) rowC[id] = s;
  } else {
    int id2 = id - BN * CL;
    const float* row = Q + (size_t)id2 * HD;
    float s = 0.f;
    for (int k = lane; k < HD; k += 32) s += row[k] * w[HD + k];
    s = wave_reduce_sum(s);
    if (lane == 0) colQ[id2] = s;
  }
}

// ---------------- K2: S = (C*w3) @ Q^T + biases (f32 WMMA 16x16x4) ----------
__global__ __launch_bounds__(256) void k2_sgemm(
    const float* __restrict__ C, const float* __restrict__ Q,
    const float* __restrict__ w, const float* __restrict__ bia,
    const float* __restrict__ rowC, const float* __restrict__ colQ,
    float* __restrict__ S) {
  __shared__ float As[16 * 68];    // (C*w3) tile: 16 rows x 64 k, pitch 68
  __shared__ float Bs[64 * 136];   // Q^T tile:   64 k   x 128 j, pitch 136
  int b  = blockIdx.x >> 6;
  int i0 = (blockIdx.x & 63) << 4;
  int wave = threadIdx.x >> 5, lane = threadIdx.x & 31;
  int l = lane & 15, hi = lane >> 4;
  int j0 = wave << 4;
  v8f acc = {};
  float bval = bia[0];
  for (int k0 = 0; k0 < HD; k0 += 64) {
    for (int e = threadIdx.x; e < 16 * 64; e += 256) {
      int r = e >> 6, k = e & 63;
      As[r * 68 + k] =
          C[(size_t)(b * CL + i0 + r) * HD + k0 + k] * w[2 * HD + k0 + k];
    }
    for (int e = threadIdx.x; e < 128 * 64; e += 256) {
      int j = e >> 6, k = e & 63;
      Bs[k * 136 + j] = Q[(size_t)(b * QL + j) * HD + k0 + k];
    }
    __syncthreads();
#pragma unroll
    for (int kk = 0; kk < 64; kk += 4) {
      v2f a, bb;
      a.x  = As[l * 68 + kk + 2 * hi];
      a.y  = As[l * 68 + kk + 2 * hi + 1];
      bb.x = Bs[(kk + 2 * hi) * 136 + j0 + l];
      bb.y = Bs[(kk + 2 * hi + 1) * 136 + j0 + l];
      acc = __builtin_amdgcn_wmma_f32_16x16x4_f32(false, a, false, bb,
                                                  (short)0, acc, false, false);
    }
    __syncthreads();
  }
  int j = j0 + l;
  float cq = colQ[b * QL + j] + bval;
#pragma unroll
  for (int r = 0; r < 8; ++r) {
    int i = i0 + r + 8 * hi;
    S[(size_t)(b * CL + i) * QL + j] = acc[r] + rowC[b * CL + i] + cq;
  }
}

// ---------------- K3: masked softmax stats (rows over j, cols over i) -------
__global__ __launch_bounds__(256) void k3_stats(
    const float* __restrict__ S, const int* __restrict__ Cm,
    const int* __restrict__ Qm, float* __restrict__ rmax,
    float* __restrict__ rsum, float* __restrict__ cmax,
    float* __restrict__ csum) {
  int wave = threadIdx.x >> 5, lane = threadIdx.x & 31;
  int id = blockIdx.x * 8 + wave;
  if (id < BN * CL) {
    int b = id / CL;
    const float* row = S + (size_t)id * QL;
    float v[4], m = NEGV;
#pragma unroll
    for (int t = 0; t < 4; ++t) {
      int j = lane + 32 * t;
      float x = Qm[b * QL + j] ? row[j] : NEGV;
      v[t] = x;
      m = fmaxf(m, x);
    }
    m = wave_reduce_max(m);
    float s = 0.f;
#pragma unroll
    for (int t = 0; t < 4; ++t) s += __expf(v[t] - m);
    s = wave_reduce_sum(s);
    if (lane == 0) { rmax[id] = m; rsum[id] = s; }
  } else {
    int id2 = id - BN * CL;
    int b = id2 >> 7, j = id2 & 127;
    const float* col = S + (size_t)(b * CL) * QL + j;
    const int* cm = Cm + b * CL;
    float m = NEGV;
    for (int i = lane; i < CL; i += 32)
      m = fmaxf(m, cm[i] ? col[(size_t)i * QL] : NEGV);
    m = wave_reduce_max(m);
    float s = 0.f;
    for (int i = lane; i < CL; i += 32)
      s += __expf((cm[i] ? col[(size_t)i * QL] : NEGV) - m);
    s = wave_reduce_sum(s);
    if (lane == 0) { cmax[id2] = m; csum[id2] = s; }
  }
}

// ---------------- K4: T = S_col^T @ C (bf16 WMMA 16x16x32) ------------------
__global__ __launch_bounds__(256) void k4_tgemm(
    const float* __restrict__ C, const float* __restrict__ S,
    const int* __restrict__ Cm, const float* __restrict__ cmax,
    const float* __restrict__ csum, float* __restrict__ T) {
  __shared__ float Ss[32 * 33];  // S_col probs: 32 k(i) x 32 j, pitch 33
  int b  = blockIdx.x >> 4;
  int jb = (blockIdx.x >> 2) & 3;
  int hb = blockIdx.x & 3;
  int j0 = jb << 5;
  int h0 = hb << 7;
  int wave = threadIdx.x >> 5, lane = threadIdx.x & 31;
  int l = lane & 15, hi = lane >> 4;
  int jt = wave >> 2;
  int n0 = h0 + ((wave & 3) << 5);
  v8f acc0 = {}, acc1 = {};
  for (int i0c = 0; i0c < CL; i0c += 32) {
    for (int e = threadIdx.x; e < 32 * 32; e += 256) {
      int k = e >> 5, j = e & 31;
      int i = i0c + k;
      float x = S[(size_t)(b * CL + i) * QL + j0 + j];
      x = Cm[b * CL + i] ? x : NEGV;
      int jj = b * QL + j0 + j;
      Ss[k * 33 + j] = __expf(x - cmax[jj]) / csum[jj];
    }
    __syncthreads();
    v16bf a, b0, b1;
#pragma unroll
    for (int e = 0; e < 16; ++e)
      a[e] = (__bf16)Ss[(e + 16 * hi) * 33 + (jt << 4) + l];
#pragma unroll
    for (int e = 0; e < 16; ++e) {
      size_t rowb = (size_t)(b * CL + i0c + 16 * hi + e) * HD;
      b0[e] = (__bf16)C[rowb + n0 + l];
      b1[e] = (__bf16)C[rowb + n0 + 16 + l];
    }
    acc0 = __builtin_amdgcn_wmma_f32_16x16x32_bf16(false, a, false, b0,
                                                   (short)0, acc0, false, false);
    acc1 = __builtin_amdgcn_wmma_f32_16x16x32_bf16(false, a, false, b1,
                                                   (short)0, acc1, false, false);
    __syncthreads();
  }
#pragma unroll
  for (int r = 0; r < 8; ++r) {
    int j = j0 + (jt << 4) + r + 8 * hi;
    size_t base = (size_t)(b * QL + j) * HD;
    T[base + n0 + l]      = acc0[r];
    T[base + n0 + 16 + l] = acc1[r];
  }
}

// ---------------- K5: A = S_row@Q, Bm = S_row@T, write concat ---------------
__global__ __launch_bounds__(256) void k5_out(
    const float* __restrict__ C, const float* __restrict__ Q,
    const float* __restrict__ Tw, const float* __restrict__ S,
    const int* __restrict__ Qm, const float* __restrict__ rmax,
    const float* __restrict__ rsum, float* __restrict__ out) {
  __shared__ float Ps[16 * 132];  // S_row probs: 16 i x 128 j, pitch 132
  int b  = blockIdx.x >> 6;
  int i0 = (blockIdx.x & 63) << 4;
  int wave = threadIdx.x >> 5, lane = threadIdx.x & 31;
  int l = lane & 15, hi = lane >> 4;
  for (int e = threadIdx.x; e < 16 * 128; e += 256) {
    int r = e >> 7, j = e & 127;
    int row = b * CL + i0 + r;
    float x = Qm[b * QL + j] ? S[(size_t)row * QL + j] : NEGV;
    Ps[r * 132 + j] = __expf(x - rmax[row]) / rsum[row];
  }
  __syncthreads();
  int h0w = wave << 6;
  v8f accA[4] = {{}, {}, {}, {}};
  v8f accB[4] = {{}, {}, {}, {}};
  for (int k0 = 0; k0 < QL; k0 += 32) {
    v16bf a;
#pragma unroll
    for (int e = 0; e < 16; ++e)
      a[e] = (__bf16)Ps[l * 132 + k0 + 16 * hi + e];
#pragma unroll
    for (int t = 0; t < 4; ++t) {
      int n0 = h0w + (t << 4);
      v16bf bq, bt;
#pragma unroll
      for (int e = 0; e < 16; ++e) {
        size_t rowb = (size_t)(b * QL + k0 + 16 * hi + e) * HD;
        bq[e] = (__bf16)Q[rowb + n0 + l];
        bt[e] = (__bf16)Tw[rowb + n0 + l];
      }
      accA[t] = __builtin_amdgcn_wmma_f32_16x16x32_bf16(
          false, a, false, bq, (short)0, accA[t], false, false);
      accB[t] = __builtin_amdgcn_wmma_f32_16x16x32_bf16(
          false, a, false, bt, (short)0, accB[t], false, false);
    }
  }
#pragma unroll
  for (int t = 0; t < 4; ++t) {
    int h = h0w + (t << 4) + l;
#pragma unroll
    for (int r = 0; r < 8; ++r) {
      int i = i0 + r + 8 * hi;
      float c  = C[(size_t)(b * CL + i) * HD + h];
      float av = accA[t][r];
      float bv = accB[t][r];
      size_t base = (size_t)(b * CL + i) * (4 * HD);
      out[base + h]          = c;
      out[base + HD + h]     = av;
      out[base + 2 * HD + h] = c * av;
      out[base + 3 * HD + h] = c * bv;
    }
  }
}

extern "C" void kernel_launch(void* const* d_in, const int* in_sizes, int n_in,
                              void* d_out, int out_size, void* d_ws,
                              size_t ws_size, hipStream_t stream) {
  (void)in_sizes; (void)n_in; (void)out_size; (void)ws_size;
  const float* C  = (const float*)d_in[0];
  const float* Q  = (const float*)d_in[1];
  const int* Cm   = (const int*)d_in[2];
  const int* Qm   = (const int*)d_in[3];
  const float* w  = (const float*)d_in[4];
  const float* bi = (const float*)d_in[5];
  float* out = (float*)d_out;

  float* p = (float*)d_ws;
  float* rowC = p; p += BN * CL;
  float* colQ = p; p += BN * QL;
  float* rmax = p; p += BN * CL;
  float* rsum = p; p += BN * CL;
  float* cmax = p; p += BN * QL;
  float* csum = p; p += BN * QL;
  float* S    = p; p += (size_t)BN * CL * QL;
  float* T    = p; p += (size_t)BN * QL * HD;

  k1_dots<<<(BN * CL + BN * QL) / 8, 256, 0, stream>>>(C, Q, w, rowC, colQ);
  k2_sgemm<<<BN * (CL / 16), 256, 0, stream>>>(C, Q, w, bi, rowC, colQ, S);
  k3_stats<<<(BN * CL + BN * QL) / 8, 256, 0, stream>>>(S, Cm, Qm, rmax, rsum,
                                                        cmax, csum);
  k4_tgemm<<<BN * 16, 256, 0, stream>>>(C, S, Cm, cmax, csum, T);
  k5_out<<<BN * (CL / 16), 256, 0, stream>>>(C, Q, T, S, Qm, rmax, rsum, out);
}
